// LSTMDecoder_22565758173554
// MI455X (gfx1250) — matrix operand order repaired and checked by reading, not compiled, not run on hardware
//
#include <hip/hip_runtime.h>
#include <hip/hip_bf16.h>
#include <stdint.h>

// ---------------------------------------------------------------------------
// LSTM decoder w/ attention for MI455X (gfx1250, wave32, WMMA).
// Dominant cost: logits = h @ Wfc^T  (M=32, N=32000, K=512) every step.
// Wfc(bf16)=32.8MB fits in 192MB L2 -> per-step weight reads are L2 hits.
// v_wmma_f32_16x16x32_bf16; one 64-thread block (2 waves) per 32x64 output
// strip: the two waves (mt=0/1) share the same B tiles via the WGP cache.
// k-loop is software pipelined (double-buffered A/B) and held in place with
// __builtin_amdgcn_sched_barrier so the compiler can't collapse it.
// ---------------------------------------------------------------------------

#define BB 32
#define TT 128
#define LL 196
#define HH 512
#define VV 32000
#define H4 2048     // 4*H
#define KCAT 1536   // H(emb) + H(ctx) + H(h)

typedef __attribute__((ext_vector_type(16))) __bf16 v16bf;
typedef __attribute__((ext_vector_type(8)))  float  v8f;

#if defined(__has_builtin)
#if __has_builtin(__builtin_amdgcn_sched_barrier)
#define SCHED_FENCE() __builtin_amdgcn_sched_barrier(0)
#endif
#endif
#ifndef SCHED_FENCE
#define SCHED_FENCE()
#endif

static __device__ __forceinline__ unsigned short f32_to_bf16(float f) {
    unsigned u = __builtin_bit_cast(unsigned, f);
    unsigned r = u + 0x7FFFu + ((u >> 16) & 1u);   // round-to-nearest-even
    return (unsigned short)(r >> 16);
}

static __device__ __forceinline__ float sigf(float x) {
    return 1.0f / (1.0f + __expf(-x));
}

static __device__ __forceinline__ v16bf nt_load(const unsigned short* p) {
    return __builtin_nontemporal_load(reinterpret_cast<const v16bf*>(p));
}

// Store element (m,k) of an A-matrix into a WMMA 16-bit A-fragment layout
// matching the B loads (contiguous k order: lanes 0-15 hold K{0..15},
// lanes 16-31 hold K{16..31}). Fragment is stored so each lane
// reads 16 contiguous bf16 (32B) per (mtile, kblock).
static __device__ __forceinline__ void store_afrag(unsigned short* frag, int KB,
                                                   int m, int k, float v) {
    int mt = m >> 4, kb = k >> 5, kk = k & 31;
    int lane = (m & 15) + ((kk >> 4) << 4);
    int pos  = kk & 15;
    frag[(((size_t)(mt * KB + kb) << 5) + lane) * 16 + pos] = f32_to_bf16(v);
}

// Core WMMA strip: acc[0..3] += Afrag(mtile mt) x W^T over K, 4 n-tiles at n0.
// W is [N,K] row-major bf16 (B[k,n] = W[n,k]); per-lane B load is one 32B
// contiguous chunk: W[n0+(lane&15) + 16j, k0 + 16*(lane>>4) .. +16].
// Software pipelined: fragments for kb+1 are issued before the WMMAs of kb;
// sched barriers keep the loads a full WMMA bundle ahead of their use.
template <int K>
static __device__ __forceinline__ void wmma_acc4(const unsigned short* __restrict__ afrag,
                                                 const unsigned short* __restrict__ wt,
                                                 int mt, int n0, int lane,
                                                 v8f acc[4]) {
    constexpr int KB = K >> 5;
    const unsigned short* arow = afrag + (((size_t)mt * KB << 5) + lane) * 16;
    const unsigned short* wb   = wt + (size_t)(n0 + (lane & 15)) * K + ((lane >> 4) << 4);

    v16bf a0 = *reinterpret_cast<const v16bf*>(arow);
    v16bf b0 = nt_load(wb + (size_t)0 * 16 * K);
    v16bf b1 = nt_load(wb + (size_t)1 * 16 * K);
    v16bf b2 = nt_load(wb + (size_t)2 * 16 * K);
    v16bf b3 = nt_load(wb + (size_t)3 * 16 * K);
#pragma unroll
    for (int kb = 0; kb < KB; ++kb) {
        const int kn = (kb + 1 < KB) ? kb + 1 : kb;      // clamp: last iter reloads
        v16bf an = *reinterpret_cast<const v16bf*>(arow + (size_t)kn * 512);
        const unsigned short* wn = wb + (kn << 5);
        v16bf bn0 = nt_load(wn + (size_t)0 * 16 * K);
        v16bf bn1 = nt_load(wn + (size_t)1 * 16 * K);
        v16bf bn2 = nt_load(wn + (size_t)2 * 16 * K);
        v16bf bn3 = nt_load(wn + (size_t)3 * 16 * K);
        SCHED_FENCE();   // loads for kb+1 stay above the WMMAs of kb
        acc[0] = __builtin_amdgcn_wmma_f32_16x16x32_bf16(false, a0, false, b0, (short)0, acc[0], false, false);
        acc[1] = __builtin_amdgcn_wmma_f32_16x16x32_bf16(false, a0, false, b1, (short)0, acc[1], false, false);
        acc[2] = __builtin_amdgcn_wmma_f32_16x16x32_bf16(false, a0, false, b2, (short)0, acc[2], false, false);
        acc[3] = __builtin_amdgcn_wmma_f32_16x16x32_bf16(false, a0, false, b3, (short)0, acc[3], false, false);
        SCHED_FENCE();   // next iteration's loads issue after this bundle
        a0 = an; b0 = bn0; b1 = bn1; b2 = bn2; b3 = bn3;
    }
}

// ---------------------------- setup kernels --------------------------------

__global__ void cvt_wfc_kernel(const float* __restrict__ wfc,
                               unsigned short* __restrict__ wfc_bf) {
    size_t i = (size_t)blockIdx.x * blockDim.x + threadIdx.x;
    if (i < (size_t)VV * HH) wfc_bf[i] = f32_to_bf16(wfc[i]);
}

__global__ void cvt_wcat_kernel(const float* __restrict__ w_ih,
                                const float* __restrict__ w_hh,
                                unsigned short* __restrict__ wcat) {
    size_t i = (size_t)blockIdx.x * blockDim.x + threadIdx.x;
    if (i >= (size_t)H4 * KCAT) return;
    int n = (int)(i / KCAT), k = (int)(i % KCAT);
    float v = (k < 2 * HH) ? w_ih[(size_t)n * (2 * HH) + k]
                           : w_hh[(size_t)n * HH + (k - 2 * HH)];
    wcat[i] = f32_to_bf16(v);
}

__global__ void init_state_kernel(const float* __restrict__ h0,
                                  const float* __restrict__ c0,
                                  float* __restrict__ hbuf,
                                  float* __restrict__ cbuf) {
    int i = blockIdx.x * blockDim.x + threadIdx.x;
    if (i < BB * HH) { hbuf[i] = h0[i]; cbuf[i] = c0[i]; }
}

// ---------------------------- per-step kernels -----------------------------

// One block per batch row: q = h@Wq^T+bq, scores, softmax, ctx; emit
// x_cat = [emb_t, ctx, h] straight into WMMA A-fragment layout (bf16).
__global__ void attn_kernel(const float* __restrict__ hbuf,
                            const float* __restrict__ features,
                            const float* __restrict__ Wq,
                            const float* __restrict__ bq,
                            const float* __restrict__ embed,
                            const int* __restrict__ inputs,
                            int t, unsigned short* __restrict__ xfrag) {
    __shared__ float sh_h[HH];
    __shared__ float sh_q[HH];
    __shared__ float sh_s[LL];
    __shared__ float sh_e[LL];
    const int b = blockIdx.x, tid = threadIdx.x, NT = blockDim.x;

    for (int k = tid; k < HH; k += NT) sh_h[k] = hbuf[b * HH + k];
    __syncthreads();

    for (int j = tid; j < HH; j += NT) {
        const float* w = Wq + (size_t)j * HH;
        float acc = bq[j];
        for (int k = 0; k < HH; ++k) acc += sh_h[k] * w[k];
        sh_q[j] = acc;
    }
    __syncthreads();

    const float* fb = features + (size_t)b * LL * HH;
    for (int l = tid; l < LL; l += NT) {
        const float* f = fb + (size_t)l * HH;
        float acc = 0.f;
        for (int k = 0; k < HH; ++k) acc += f[k] * sh_q[k];
        sh_s[l] = acc;
    }
    __syncthreads();

    float mx = -3.4e38f;                       // redundant per-thread reduce (L=196)
    for (int l = 0; l < LL; ++l) mx = fmaxf(mx, sh_s[l]);
    for (int l = tid; l < LL; l += NT) sh_e[l] = __expf(sh_s[l] - mx);
    __syncthreads();
    float sm = 0.f;
    for (int l = 0; l < LL; ++l) sm += sh_e[l];
    const float inv = 1.0f / sm;

    for (int k = tid; k < HH; k += NT) {       // ctx, coalesced over k per l
        float acc = 0.f;
        for (int l = 0; l < LL; ++l) acc += sh_e[l] * fb[(size_t)l * HH + k];
        store_afrag(xfrag, KCAT >> 5, b, HH + k, acc * inv);
    }
    const int tok = inputs[b * TT + t];
    const float* eb = embed + (size_t)tok * HH;
    for (int k = tid; k < HH; k += NT) store_afrag(xfrag, KCAT >> 5, b, k, eb[k]);
    for (int k = tid; k < HH; k += NT) store_afrag(xfrag, KCAT >> 5, b, 2 * HH + k, sh_h[k]);
}

// gates[32,2048] = x_cat(bf16) @ Wcat^T + b_ih + b_hh
// 64-thread blocks: wave0 -> mt=0, wave1 -> mt=1, shared B tiles (n-strip).
__global__ void gates_kernel(const unsigned short* __restrict__ xfrag,
                             const unsigned short* __restrict__ wcat,
                             const float* __restrict__ b_ih,
                             const float* __restrict__ b_hh,
                             float* __restrict__ gates) {
    const int lane = threadIdx.x & 31;
    const int mt   = threadIdx.x >> 5;
    const int n0   = blockIdx.x * 64;
    v8f acc[4] = {};
    wmma_acc4<KCAT>(xfrag, wcat, mt, n0, lane, acc);
#pragma unroll
    for (int j = 0; j < 4; ++j) {
        const int col = n0 + j * 16 + (lane & 15);
        const float bias = b_ih[col] + b_hh[col];
#pragma unroll
        for (int r = 0; r < 8; ++r) {
            const int m = mt * 16 + r + ((lane >> 4) << 3);
            gates[(size_t)m * H4 + col] = acc[j][r] + bias;
        }
    }
}

// LSTM pointwise; emits h as f32 (next-step attention) and bf16 A-fragment.
__global__ void lstm_kernel(const float* __restrict__ gates,
                            float* __restrict__ hbuf,
                            float* __restrict__ cbuf,
                            unsigned short* __restrict__ hfrag) {
    const int idx = blockIdx.x * blockDim.x + threadIdx.x;
    if (idx >= BB * HH) return;
    const int b = idx >> 9, n = idx & (HH - 1);
    const float* g = gates + (size_t)b * H4;
    const float gi = g[n], gf = g[HH + n], gg = g[2 * HH + n], go = g[3 * HH + n];
    const float c = sigf(gf) * cbuf[idx] + sigf(gi) * tanhf(gg);
    const float h = sigf(go) * tanhf(c);
    cbuf[idx] = c;
    hbuf[idx] = h;
    store_afrag(hfrag, HH >> 5, b, n, h);
}

// logits[32,32000] = h(bf16) @ Wfc^T + bfc
// 500 blocks x 64 threads: one block per 64-wide n-strip, 2 waves = 2 m-tiles.
__global__ void logits_kernel(const unsigned short* __restrict__ hfrag,
                              const unsigned short* __restrict__ wfc_bf,
                              const float* __restrict__ bfc,
                              float* __restrict__ out, int t) {
    const int lane = threadIdx.x & 31;
    const int mt   = threadIdx.x >> 5;
    const int n0   = blockIdx.x * 64;
    v8f acc[4] = {};
    wmma_acc4<HH>(hfrag, wfc_bf, mt, n0, lane, acc);
#pragma unroll
    for (int j = 0; j < 4; ++j) {
        const int col = n0 + j * 16 + (lane & 15);
        const float bias = bfc[col];
#pragma unroll
        for (int r = 0; r < 8; ++r) {
            const int m = mt * 16 + r + ((lane >> 4) << 3);
            out[((size_t)m * TT + t) * VV + col] = acc[j][r] + bias;
        }
    }
}

__global__ void finalize_kernel(const float* __restrict__ hbuf,
                                const float* __restrict__ cbuf,
                                float* __restrict__ out_tail) {
    int i = blockIdx.x * blockDim.x + threadIdx.x;
    if (i < BB * HH) { out_tail[i] = hbuf[i]; out_tail[BB * HH + i] = cbuf[i]; }
}

// ------------------------------- launcher ----------------------------------

extern "C" void kernel_launch(void* const* d_in, const int* in_sizes, int n_in,
                              void* d_out, int out_size, void* d_ws, size_t ws_size,
                              hipStream_t stream) {
    const int*   inputs   = (const int*)  d_in[0];
    const float* h0       = (const float*)d_in[1];
    const float* c0       = (const float*)d_in[2];
    const float* features = (const float*)d_in[3];
    const float* embed    = (const float*)d_in[4];
    const float* Wq       = (const float*)d_in[5];
    const float* bq       = (const float*)d_in[6];
    const float* W_ih     = (const float*)d_in[7];
    const float* b_ih     = (const float*)d_in[8];
    const float* W_hh     = (const float*)d_in[9];
    const float* b_hh     = (const float*)d_in[10];
    const float* Wfc      = (const float*)d_in[11];
    const float* bfc      = (const float*)d_in[12];
    float* out = (float*)d_out;

    // workspace layout (all 256B aligned)
    char* ws = (char*)d_ws;
    size_t off = 0;
    unsigned short* wfc_bf = (unsigned short*)(ws + off); off += (size_t)VV * HH * 2;   // 32.77 MB
    unsigned short* wcat   = (unsigned short*)(ws + off); off += (size_t)H4 * KCAT * 2; // 6.29 MB
    float* hbuf            = (float*)(ws + off);          off += (size_t)BB * HH * 4;
    float* cbuf            = (float*)(ws + off);          off += (size_t)BB * HH * 4;
    unsigned short* hfrag  = (unsigned short*)(ws + off); off += (size_t)BB * HH * 2;
    unsigned short* xfrag  = (unsigned short*)(ws + off); off += (size_t)BB * KCAT * 2;
    float* gates           = (float*)(ws + off);          off += (size_t)BB * H4 * 4;

    // one-time conversions (re-run every call: deterministic & idempotent)
    cvt_wfc_kernel <<<((size_t)VV * HH + 255) / 256, 256, 0, stream>>>(Wfc, wfc_bf);
    cvt_wcat_kernel<<<((size_t)H4 * KCAT + 255) / 256, 256, 0, stream>>>(W_ih, W_hh, wcat);
    init_state_kernel<<<(BB * HH + 255) / 256, 256, 0, stream>>>(h0, c0, hbuf, cbuf);

    for (int t = 0; t < TT; ++t) {
        attn_kernel  <<<BB, 256, 0, stream>>>(hbuf, features, Wq, bq, embed, inputs, t, xfrag);
        gates_kernel <<<H4 / 64, 64, 0, stream>>>(xfrag, wcat, b_ih, b_hh, gates);   // 32 blocks
        lstm_kernel  <<<(BB * HH + 255) / 256, 256, 0, stream>>>(gates, hbuf, cbuf, hfrag);
        logits_kernel<<<VV / 64, 64, 0, stream>>>(hfrag, wfc_bf, bfc, out, t);       // 500 blocks
    }
    finalize_kernel<<<(BB * HH + 255) / 256, 256, 0, stream>>>(hbuf, cbuf, out + (size_t)BB * TT * VV);
    (void)in_sizes; (void)n_in; (void)out_size; (void)ws_size;
}
